// AbsTopKSAE_8753143349771
// MI455X (gfx1250) — matrix operand order repaired and checked by reading, not compiled, not run on hardware
//
#include <hip/hip_runtime.h>
#include <hip/hip_bf16.h>

#define B_DIM 4096
#define D_DIM 768
#define F_DIM 24576
#define TOPK  72
#define L1C   1e-3f
#define FULL_PARTS ((D_DIM / 64) * (B_DIM / 128))   // 12 * 32 = 384

typedef __attribute__((ext_vector_type(16))) __bf16 v16bf;
typedef __attribute__((ext_vector_type(8)))  __bf16 v8bf;
typedef __attribute__((ext_vector_type(8)))  float  v8f;
typedef __attribute__((ext_vector_type(4)))  float  v4f;

static __device__ __forceinline__ v8f wmma_bf16(v16bf a, v16bf b, v8f c) {
  // D = A(16x32) * B(32x16) + C, f32 accumulate
  return __builtin_amdgcn_wmma_f32_16x16x32_bf16(false, a, false, b,
                                                 (short)0, c, false, false);
}

static __device__ __forceinline__ v16bf make_afrag(const __bf16* ap) {
  v8bf lo = *(const v8bf*)ap;          // K = khalf*8 + 0..7
  v8bf hi = *(const v8bf*)(ap + 16);   // K = 16 + khalf*8 + 0..7
  v16bf a;
#pragma unroll
  for (int e = 0; e < 8; ++e) { a[e] = lo[e]; a[e + 8] = hi[e]; }
  return a;
}

// ---------------------------------------------------------------------------
// Kernel 1: latents[B,F] = (x - b_dec) @ W_enc     (bf16 WMMA, f32 accum)
// block tile 128(M) x 128(N), BK=32, 256 threads = 8 waves (4x2 wave grid),
// each wave owns 32x64 = 2x4 accumulators. Register-double-buffered staging.
// ---------------------------------------------------------------------------
__global__ __launch_bounds__(256) void encode_gemm(
    const float* __restrict__ x, const float* __restrict__ Wenc,
    const float* __restrict__ bdec, float* __restrict__ latents) {
  __shared__ alignas(64) __bf16 As[128][32];   // A[m][k]
  __shared__ alignas(64) __bf16 Bs[128][32];   // B^T[n][k]
  const int t    = threadIdx.x;
  const int lane = t & 31, wave = t >> 5;
  const int wm = wave & 3, wn = wave >> 2;          // 4 x 2 wave grid
  const int rowBase = blockIdx.y * 128;
  const int colBase = blockIdx.x * 128;
  const int mlo = lane & 15, khalf = lane >> 4;

  // branch-free per-thread staging slots (4 float4 for A, 4 float4 for B)
  int rA[4], kgA[4], kB[4], ngB[4];
#pragma unroll
  for (int c = 0; c < 4; ++c) {
    int ia = t + 256 * c;
    rA[c]  = ia >> 3; kgA[c] = (ia & 7) << 2;    // A: 128 rows x 8 float4
    kB[c]  = ia >> 5; ngB[c] = (ia & 31) << 2;   // B: 32 rows x 32 float4
  }

  v8f acc[2][4] = {};
  v4f ax[4], bw[4];

  // prologue: load tile kb=0 into registers
#pragma unroll
  for (int c = 0; c < 4; ++c) {
    ax[c] = *(const v4f*)(x + (size_t)(rowBase + rA[c]) * D_DIM + kgA[c]);
    bw[c] = *(const v4f*)(Wenc + (size_t)kB[c] * F_DIM + colBase + ngB[c]);
  }

  for (int kb = 0; kb < D_DIM; kb += 32) {
    __syncthreads();   // previous iteration's ds_loads complete
    // convert + store current tile to LDS
#pragma unroll
    for (int c = 0; c < 4; ++c) {
      v4f bv = *(const v4f*)(bdec + kb + kgA[c]);
      As[rA[c]][kgA[c] + 0] = (__bf16)(ax[c].x - bv.x);
      As[rA[c]][kgA[c] + 1] = (__bf16)(ax[c].y - bv.y);
      As[rA[c]][kgA[c] + 2] = (__bf16)(ax[c].z - bv.z);
      As[rA[c]][kgA[c] + 3] = (__bf16)(ax[c].w - bv.w);
      Bs[ngB[c] + 0][kB[c]] = (__bf16)bw[c].x;
      Bs[ngB[c] + 1][kB[c]] = (__bf16)bw[c].y;
      Bs[ngB[c] + 2][kB[c]] = (__bf16)bw[c].z;
      Bs[ngB[c] + 3][kB[c]] = (__bf16)bw[c].w;
    }
    __syncthreads();   // LDS tile ready
    // prefetch next tile while WMMAs run
    if (kb + 32 < D_DIM) {
      const int kn = kb + 32;
#pragma unroll
      for (int c = 0; c < 4; ++c) {
        ax[c] = *(const v4f*)(x + (size_t)(rowBase + rA[c]) * D_DIM + kn + kgA[c]);
        bw[c] = *(const v4f*)(Wenc + (size_t)(kn + kB[c]) * F_DIM + colBase + ngB[c]);
      }
    }
    v16bf afrag[2];
#pragma unroll
    for (int i = 0; i < 2; ++i)
      afrag[i] = make_afrag(&As[wm * 32 + i * 16 + mlo][khalf * 8]);
#pragma unroll
    for (int j = 0; j < 4; ++j) {
      v16bf bfrag = make_afrag(&Bs[wn * 64 + j * 16 + mlo][khalf * 8]);
#pragma unroll
      for (int i = 0; i < 2; ++i) acc[i][j] = wmma_bf16(afrag[i], bfrag, acc[i][j]);
    }
  }
  // epilogue: C/D layout -> row = 8*(lane>=16)+vgpr, col = lane&15
  const int mOff = (lane >> 4) * 8, nIdx = lane & 15;
#pragma unroll
  for (int i = 0; i < 2; ++i)
#pragma unroll
    for (int j = 0; j < 4; ++j)
#pragma unroll
      for (int r = 0; r < 8; ++r) {
        int row = rowBase + wm * 32 + i * 16 + mOff + r;
        int col = colBase + wn * 64 + j * 16 + nIdx;
        latents[(size_t)row * F_DIM + col] = acc[i][j][r];
      }
}

// ---------------------------------------------------------------------------
// Kernel 2: l2_full partials of (latents @ W_dec + b_dec - x)^2
// block tile 128(M) x 64(N), BK=32 -> 384 blocks (better fill; W_dec and hot
// latents tiles are L2-resident so the extra N-splitting adds no HBM traffic).
// 8 waves in 4x2 grid, each wave 32x32 = 2x2 accumulators.
// ---------------------------------------------------------------------------
__global__ __launch_bounds__(256) void full_gemm_loss(
    const float* __restrict__ latents, const float* __restrict__ Wdec,
    const float* __restrict__ bdec, const float* __restrict__ x,
    float* __restrict__ partials) {
  __shared__ alignas(64) __bf16 As[128][32];
  __shared__ alignas(64) __bf16 Bs[64][32];
  __shared__ float red[256];
  const int t    = threadIdx.x;
  const int lane = t & 31, wave = t >> 5;
  const int wm = wave & 3, wn = wave >> 2;
  const int rowBase = blockIdx.y * 128;
  const int colBase = blockIdx.x * 64;
  const int mlo = lane & 15, khalf = lane >> 4;

  int rA[4], kgA[4], kB[2], ngB[2];
#pragma unroll
  for (int c = 0; c < 4; ++c) {
    int ia = t + 256 * c;
    rA[c] = ia >> 3; kgA[c] = (ia & 7) << 2;     // A: 128 rows x 8 float4
  }
#pragma unroll
  for (int c = 0; c < 2; ++c) {
    int ib = t + 256 * c;
    kB[c] = ib >> 4; ngB[c] = (ib & 15) << 2;    // B: 32 rows x 16 float4
  }

  v8f acc[2][2] = {};
  v4f ax[4], bw[2];

#pragma unroll
  for (int c = 0; c < 4; ++c)
    ax[c] = *(const v4f*)(latents + (size_t)(rowBase + rA[c]) * F_DIM + kgA[c]);
#pragma unroll
  for (int c = 0; c < 2; ++c)
    bw[c] = *(const v4f*)(Wdec + (size_t)kB[c] * D_DIM + colBase + ngB[c]);

  for (int kb = 0; kb < F_DIM; kb += 32) {
    __syncthreads();
#pragma unroll
    for (int c = 0; c < 4; ++c) {
      As[rA[c]][kgA[c] + 0] = (__bf16)ax[c].x;
      As[rA[c]][kgA[c] + 1] = (__bf16)ax[c].y;
      As[rA[c]][kgA[c] + 2] = (__bf16)ax[c].z;
      As[rA[c]][kgA[c] + 3] = (__bf16)ax[c].w;
    }
#pragma unroll
    for (int c = 0; c < 2; ++c) {
      Bs[ngB[c] + 0][kB[c]] = (__bf16)bw[c].x;
      Bs[ngB[c] + 1][kB[c]] = (__bf16)bw[c].y;
      Bs[ngB[c] + 2][kB[c]] = (__bf16)bw[c].z;
      Bs[ngB[c] + 3][kB[c]] = (__bf16)bw[c].w;
    }
    __syncthreads();
    if (kb + 32 < F_DIM) {
      const int kn = kb + 32;
#pragma unroll
      for (int c = 0; c < 4; ++c)
        ax[c] = *(const v4f*)(latents + (size_t)(rowBase + rA[c]) * F_DIM + kn + kgA[c]);
#pragma unroll
      for (int c = 0; c < 2; ++c)
        bw[c] = *(const v4f*)(Wdec + (size_t)(kn + kB[c]) * D_DIM + colBase + ngB[c]);
    }
    v16bf afrag[2];
#pragma unroll
    for (int i = 0; i < 2; ++i)
      afrag[i] = make_afrag(&As[wm * 32 + i * 16 + mlo][khalf * 8]);
#pragma unroll
    for (int j = 0; j < 2; ++j) {
      v16bf bfrag = make_afrag(&Bs[wn * 32 + j * 16 + mlo][khalf * 8]);
#pragma unroll
      for (int i = 0; i < 2; ++i) acc[i][j] = wmma_bf16(afrag[i], bfrag, acc[i][j]);
    }
  }
  const int mOff = (lane >> 4) * 8, nIdx = lane & 15;
  float lsum = 0.f;
#pragma unroll
  for (int i = 0; i < 2; ++i)
#pragma unroll
    for (int j = 0; j < 2; ++j)
#pragma unroll
      for (int r = 0; r < 8; ++r) {
        int row = rowBase + wm * 32 + i * 16 + mOff + r;
        int col = colBase + wn * 32 + j * 16 + nIdx;
        float v = acc[i][j][r] + bdec[col];
        float d = v - x[(size_t)row * D_DIM + col];
        lsum += d * d;
      }
  red[t] = lsum;
  __syncthreads();
  for (int s = 128; s > 0; s >>= 1) {
    if (t < s) red[t] += red[t + s];
    __syncthreads();
  }
  if (t == 0) partials[blockIdx.y * gridDim.x + blockIdx.x] = red[0];
}

// ---------------------------------------------------------------------------
// Kernel 3: AbsTopK per row. One block per row, 256 threads, 96 contiguous
// elements/thread. bf16 |v| keys cached in 48KB LDS; binary-search threshold,
// deterministic index-ordered tie break via serial prefix scans.
// Zeroes non-kept latents in place; emits compact (idx,val) list + l1/l0.
// ---------------------------------------------------------------------------
__global__ __launch_bounds__(256) void abstopk(
    float* __restrict__ acts, float* __restrict__ tv, int* __restrict__ ti,
    float* __restrict__ rowL1, float* __restrict__ rowL0) {
  __shared__ __bf16 keys[F_DIM];     // 48 KB
  __shared__ float  sred[256];
  __shared__ int    sint[256];
  __shared__ int    sscan[257];
  const int t = threadIdx.x;
  const size_t rbase = (size_t)blockIdx.x * F_DIM;
  const int c0 = t * 96;

  float vmax = 0.f;
  for (int i = t; i < F_DIM; i += 256) {
    float a = fabsf(acts[rbase + i]);
    keys[i] = (__bf16)a;
    vmax = fmaxf(vmax, a);
  }
  sred[t] = vmax;
  __syncthreads();
  for (int s = 128; s > 0; s >>= 1) {
    if (t < s) sred[t] = fmaxf(sred[t], sred[t + s]);
    __syncthreads();
  }
  float hi = sred[0], lo = 0.f;   // invariant: count(>hi) <= K < count(>lo)
  __syncthreads();

  for (int it = 0; it < 26; ++it) {
    float mid = 0.5f * (lo + hi);
    int cnt = 0;
#pragma unroll 8
    for (int i = 0; i < 96; ++i) cnt += ((float)keys[c0 + i] > mid) ? 1 : 0;
    sred[t] = (float)cnt;
    __syncthreads();
    for (int s = 128; s > 0; s >>= 1) {
      if (t < s) sred[t] += sred[t + s];
      __syncthreads();
    }
    int total = (int)sred[0];
    __syncthreads();
    if (total > TOPK) lo = mid; else hi = mid;
  }

  // classify: strictly-greater vs boundary candidates in (lo, hi]
  int ngt = 0, ncand = 0;
#pragma unroll 8
  for (int i = 0; i < 96; ++i) {
    float k = (float)keys[c0 + i];
    bool g = (k > hi);
    ngt += g ? 1 : 0;
    ncand += (!g && (k > lo)) ? 1 : 0;
  }
  sint[t] = ngt;
  __syncthreads();
  if (t == 0) { int a = 0; for (int i = 0; i < 256; ++i) { sscan[i] = a; a += sint[i]; } sscan[256] = a; }
  __syncthreads();
  const int gtBefore = sscan[t];
  const int totalGt  = sscan[256];
  __syncthreads();
  sint[t] = ncand;
  __syncthreads();
  if (t == 0) { int a = 0; for (int i = 0; i < 256; ++i) { sscan[i] = a; a += sint[i]; } sscan[256] = a; }
  __syncthreads();
  const int candBefore = sscan[t];
  const int need = TOPK - totalGt;   // >=0 by invariant

  int lgt = 0, lcand = 0, lkept = 0;
  float l1 = 0.f;
  for (int i2 = 0; i2 < 96; ++i2) {
    int i = c0 + i2;
    float v = acts[rbase + i];
    float k = (float)keys[i];
    bool isGt = (k > hi);
    bool isCand = (!isGt) && (k > lo);
    bool kept = isGt || (isCand && ((candBefore + lcand) < need));
    if (kept) {
      int pos = gtBefore + lgt + min(candBefore + lcand, need);
      tv[(size_t)blockIdx.x * TOPK + pos] = v;
      ti[(size_t)blockIdx.x * TOPK + pos] = i;
      l1 += fabsf(v);
      ++lkept;
    } else {
      acts[rbase + i] = 0.0f;
    }
    lgt += isGt ? 1 : 0;
    lcand += isCand ? 1 : 0;
  }
  sred[t] = l1;
  sint[t] = lkept;
  __syncthreads();
  for (int s = 128; s > 0; s >>= 1) {
    if (t < s) { sred[t] += sred[t + s]; sint[t] += sint[t + s]; }
    __syncthreads();
  }
  if (t == 0) {
    rowL1[blockIdx.x] = sred[0];
    rowL0[blockIdx.x] = (float)sint[0];
    for (int p = sint[0]; p < TOPK; ++p) {   // safety fill (normally empty)
      tv[(size_t)blockIdx.x * TOPK + p] = 0.f;
      ti[(size_t)blockIdx.x * TOPK + p] = 0;
    }
  }
}

// ---------------------------------------------------------------------------
// Kernel 4: sparse decode. One block per row; 72 gathered W_dec rows are
// L2-resident (W_dec = 75MB < 192MB L2). Writes x_reconstruct + l2 partial.
// ---------------------------------------------------------------------------
__global__ __launch_bounds__(256) void sparse_decode(
    const float* __restrict__ tv, const int* __restrict__ ti,
    const float* __restrict__ bdec, const float* __restrict__ x,
    const float* __restrict__ Wdec, float* __restrict__ recon,
    float* __restrict__ rowL2) {
  __shared__ float sval[TOPK];
  __shared__ int   sidx[TOPK];
  __shared__ float red[256];
  const int t = threadIdx.x;
  const int r = blockIdx.x;
  if (t < TOPK) {
    sval[t] = tv[(size_t)r * TOPK + t];
    sidx[t] = ti[(size_t)r * TOPK + t];
  }
  __syncthreads();
  float acc0 = bdec[t], acc1 = bdec[t + 256], acc2 = bdec[t + 512];
#pragma unroll 4
  for (int k = 0; k < TOPK; ++k) {
    float v = sval[k];
    const float* wr = Wdec + (size_t)sidx[k] * D_DIM;
    acc0 = fmaf(v, wr[t], acc0);
    acc1 = fmaf(v, wr[t + 256], acc1);
    acc2 = fmaf(v, wr[t + 512], acc2);
  }
  const size_t ob = (size_t)r * D_DIM;
  recon[ob + t] = acc0;
  recon[ob + t + 256] = acc1;
  recon[ob + t + 512] = acc2;
  float d0 = acc0 - x[ob + t];
  float d1 = acc1 - x[ob + t + 256];
  float d2 = acc2 - x[ob + t + 512];
  red[t] = d0 * d0 + d1 * d1 + d2 * d2;
  __syncthreads();
  for (int s = 128; s > 0; s >>= 1) {
    if (t < s) red[t] += red[t + s];
    __syncthreads();
  }
  if (t == 0) rowL2[r] = red[0];
}

// ---------------------------------------------------------------------------
// Kernel 5: deterministic final reductions -> 5 scalars.
// ---------------------------------------------------------------------------
__global__ __launch_bounds__(256) void finalize(
    const float* __restrict__ rowL2, const float* __restrict__ fullPart,
    const float* __restrict__ rowL1, const float* __restrict__ rowL0,
    float* __restrict__ outScalars) {
  __shared__ float red[256];
  const int t = threadIdx.x;
  float s2 = 0.f, sf = 0.f, s1 = 0.f, s0 = 0.f;
  for (int i = t; i < B_DIM; i += 256) {
    s2 += rowL2[i];
    s1 += rowL1[i];
    s0 += rowL0[i];
  }
  for (int i = t; i < FULL_PARTS; i += 256) sf += fullPart[i];
  red[t] = s2; __syncthreads();
  for (int s = 128; s > 0; s >>= 1) { if (t < s) red[t] += red[t + s]; __syncthreads(); }
  float sumL2 = red[0]; __syncthreads();
  red[t] = sf; __syncthreads();
  for (int s = 128; s > 0; s >>= 1) { if (t < s) red[t] += red[t + s]; __syncthreads(); }
  float sumFull = red[0]; __syncthreads();
  red[t] = s1; __syncthreads();
  for (int s = 128; s > 0; s >>= 1) { if (t < s) red[t] += red[t + s]; __syncthreads(); }
  float sumL1 = red[0]; __syncthreads();
  red[t] = s0; __syncthreads();
  for (int s = 128; s > 0; s >>= 1) { if (t < s) red[t] += red[t + s]; __syncthreads(); }
  float sumL0 = red[0];
  if (t == 0) {
    const float nBD = (float)((size_t)B_DIM * D_DIM);
    float l2 = sumL2 / nBD;
    float l2full = sumFull / nBD;
    float l1n = sumL1 / (float)B_DIM;
    float l0n = sumL0 / (float)B_DIM;
    outScalars[0] = l2 + L1C * l1n;  // loss
    outScalars[1] = l2;              // l2_loss
    outScalars[2] = l2full;          // l2_loss_full_raw
    outScalars[3] = l1n;             // l1_norm
    outScalars[4] = l0n;             // l0_norm
  }
}

// ---------------------------------------------------------------------------
extern "C" void kernel_launch(void* const* d_in, const int* in_sizes, int n_in,
                              void* d_out, int out_size, void* d_ws, size_t ws_size,
                              hipStream_t stream) {
  (void)in_sizes; (void)n_in; (void)out_size; (void)ws_size;
  const float* x    = (const float*)d_in[0];
  const float* Wenc = (const float*)d_in[1];
  const float* Wdec = (const float*)d_in[2];
  const float* bdec = (const float*)d_in[3];

  float* recon   = (float*)d_out;                              // [B, D]
  float* acts    = recon + (size_t)B_DIM * D_DIM;              // [B, F] (latents -> topk'd in place)
  float* scalars = acts + (size_t)B_DIM * F_DIM;               // 5 scalars

  float* wsf      = (float*)d_ws;
  float* tv       = wsf;                                       // 4096*72 vals
  int*   ti       = (int*)(wsf + (size_t)B_DIM * TOPK);        // 4096*72 idx
  float* rowL1    = wsf + 2 * (size_t)B_DIM * TOPK;            // 4096
  float* rowL0    = rowL1 + B_DIM;                             // 4096
  float* rowL2    = rowL0 + B_DIM;                             // 4096
  float* fullPart = rowL2 + B_DIM;                             // 384

  encode_gemm<<<dim3(F_DIM / 128, B_DIM / 128), 256, 0, stream>>>(x, Wenc, bdec, acts);
  full_gemm_loss<<<dim3(D_DIM / 64, B_DIM / 128), 256, 0, stream>>>(acts, Wdec, bdec, x, fullPart);
  abstopk<<<B_DIM, 256, 0, stream>>>(acts, tv, ti, rowL1, rowL0);
  sparse_decode<<<B_DIM, 256, 0, stream>>>(tv, ti, bdec, x, Wdec, recon, rowL2);
  finalize<<<1, 256, 0, stream>>>(rowL2, fullPart, rowL1, rowL0, scalars);
}
